// SwinT_76811195122349
// MI455X (gfx1250) — compile-verified, matrix operand order checked by reading, not run on hardware
//
#include <hip/hip_runtime.h>

// ===========================================================================
// Swin-T forward for MI455X (gfx1250, wave32).
//  - All linear projections via bf16 WMMA GEMM (v_wmma_f32_16x16x32_bf16,
//    f32 accumulate), 64x64 block tile, 8 waves each owning 32x16 (2 WMMAs).
//  - Weights pre-transposed+converted per GEMM into bf16 [Npad][K]; the GEMM
//    hot loop stages BOTH operands with a single unconditional 16B
//    global->LDS transfer per thread. If the toolchain exposes the gfx1250
//    async-DMA builtins, staging uses GLOBAL_LOAD_ASYNC_TO_LDS_B128 tracked
//    by ASYNCcnt; otherwise a b128 vector copy. Speculative prefetch of the
//    next K-tile either way (global_prefetch_b8).
//  - Stage-0 patch-embed folded into stage-0 merge weight (exact; K 1536->64).
//  - 49x49 windowed attention (dh=32 everywhere) in f32 VALU (only ~4 GFLOP
//    total; precision-sensitive softmax), one block per (batch,window,head).
//  - Residual add + cyclic roll fused into the proj-GEMM epilogue.
// ===========================================================================

typedef __attribute__((ext_vector_type(16))) __bf16 v16bf;
typedef __attribute__((ext_vector_type(8)))  float  v8f;

#define BATCH 32

#if defined(__gfx1250__) && __has_builtin(__builtin_amdgcn_global_load_async_to_lds_b128) && __has_builtin(__builtin_amdgcn_s_wait_asynccnt)
#define USE_ASYNC_LDS 1
// The builtin takes generic pointers to a 16B int vector (per clang's
// diagnostic: '__attribute__((__vector_size__(4 * sizeof(int)))) int *').
typedef int v4i_async __attribute__((vector_size(4 * sizeof(int))));
#else
#define USE_ASYNC_LDS 0
#endif

static __device__ __forceinline__ __bf16 tobf(float f) { return (__bf16)f; }

// ---------------------------------------------------------------------------
// Fold patch-embed (3->96) into stage-0 merge weight; emit directly as bf16
// transposed [Npad=128][K=64] (zero-padded), plus combined f32 bias:
//   Wc[o][(w*4+h)*3+ch] = sum_c emb_w[ch][c] * m_w[(w*4+h)*96+c][o]
//   bc[o] = m_b[o] + sum_{wh,c} emb_b[c] * m_w[wh*96+c][o]
// ---------------------------------------------------------------------------
__global__ void prep_w0_kernel(const float* __restrict__ emb_w, const float* __restrict__ emb_b,
                               const float* __restrict__ m_w, const float* __restrict__ m_b,
                               __bf16* __restrict__ WcT, float* __restrict__ bc) {
  int t = blockIdx.x * blockDim.x + threadIdx.x;
  if (t < 128 * 64) {
    int kk = t % 64, o = t / 64;
    float s = 0.f;
    if (o < 96 && kk < 48) {
      int ch = kk % 3, wh = kk / 3;
      for (int c = 0; c < 96; ++c) s += emb_w[ch * 96 + c] * m_w[(wh * 96 + c) * 96 + o];
    }
    WcT[t] = tobf(s);
  }
  if (t < 96) {
    float s = m_b[t];
    for (int wh = 0; wh < 16; ++wh)
      for (int c = 0; c < 96; ++c) s += emb_b[c] * m_w[(wh * 96 + c) * 96 + t];
    bc[t] = s;
  }
}

// Weight transpose+convert: W (nw groups of [K][N] f32, row-major) ->
// WT (nw groups of [Npad][K] bf16), rows n>=N zero. Read-coalesced in n.
__global__ void wtrans_kernel(const float* __restrict__ W, __bf16* __restrict__ WT,
                              int K, int N, int Npad, int nw) {
  long per = (long)K * Npad;
  long t = (long)blockIdx.x * 256 + threadIdx.x;
  if (t >= per * nw) return;
  int widx = (int)(t / per);
  long r = t % per;
  int k = (int)(r / Npad), n = (int)(r % Npad);
  float v = (n < N) ? W[(long)widx * K * N + (long)k * N + n] : 0.f;
  WT[(long)widx * per + (long)n * K + k] = tobf(v);
}

// A0[b,p1,p2, col] ; col<48: feats[b, ch, 4*p1+w, 4*p2+h], col in [48,64): 0
__global__ void gather0_kernel(const float* __restrict__ feats, __bf16* __restrict__ A0) {
  long t = (long)blockIdx.x * 256 + threadIdx.x;
  const long total = (long)BATCH * 56 * 56 * 64;
  if (t >= total) return;
  int col = (int)(t % 64);
  long rowi = t / 64;
  float v = 0.f;
  if (col < 48) {
    int p2 = (int)(rowi % 56);
    long r2 = rowi / 56;
    int p1 = (int)(r2 % 56);
    int b  = (int)(r2 / 56);
    int ch = col % 3, wh = col / 3;
    int w = wh / 4, h = wh % 4;
    int y = p1 * 4 + w, x = p2 * 4 + h;
    v = feats[(((long)b * 3 + ch) * 224 + y) * 224 + x];
  }
  A0[t] = tobf(v);
}

// Am[b,p1,p2,(w*2+h)*d+c] = x[b, 2p1+w, 2p2+h, c]  (bf16)
__global__ void gather_merge_kernel(const float* __restrict__ x, __bf16* __restrict__ Am,
                                    int side, int d) {
  int s2 = side / 2;
  int K = 4 * d;
  long total = (long)BATCH * s2 * s2 * K;
  long t = (long)blockIdx.x * 256 + threadIdx.x;
  if (t >= total) return;
  int col = (int)(t % K);
  long rowi = t / K;
  int p2 = (int)(rowi % s2);
  long r2 = rowi / s2;
  int p1 = (int)(r2 % s2);
  int b  = (int)(r2 / s2);
  int c = col % d, wh = col / d;
  int w = wh >> 1, h = wh & 1;
  Am[t] = tobf(x[(((long)b * side + (p1 * 2 + w)) * side + (p2 * 2 + h)) * d + c]);
}

// ---------------------------------------------------------------------------
// WMMA GEMM core, 64x64 tile. Preconditions: K % 32 == 0; A has >=64 rows of
// slack past rowBase; WT is [Npad][K] bf16 with Npad >= colBase+64 (+slack).
// ---------------------------------------------------------------------------
static __device__ __forceinline__ void gemm_core(const __bf16* __restrict__ Ag,
                                                 const __bf16* __restrict__ WTg,
                                                 int K, int rowBase, int colBase,
                                                 __bf16* lA, __bf16* lBT,
                                                 v8f& acc0, v8f& acc1) {
  const int tid  = threadIdx.x;
  const int lane = tid & 31;
  const int wv   = tid >> 5;
  const int wm   = wv & 1;            // 0..1  (M direction, 32 rows each)
  const int wn   = wv >> 1;           // 0..3  (N direction, 16 cols each)
  const int row  = lane & 15;
  const int kb   = (lane >> 4) * 8;   // K-half selected by lane[4]

  // Staging: one 16B (8 x bf16) transfer per thread for each operand.
  const int am = tid >> 2;            // A tile row 0..63
  const int ak = (tid & 3) * 8;       // 0,8,16,24
  const int bn  = tid & 63;           // W tile col (n) 0..63
  const int bk0 = (tid >> 6) * 8;     // 0,8,16,24

  #pragma unroll
  for (int i = 0; i < 8; ++i) { acc0[i] = 0.f; acc1[i] = 0.f; }

  const __bf16* aptr = Ag + (long)(rowBase + am) * K + ak;
  const __bf16* wptr = WTg + (long)(colBase + bn) * K + bk0;

  for (int k0 = 0; k0 < K; k0 += 32) {
    // speculative prefetch of next K-tile (dropped by HW if past allocation)
    __builtin_prefetch(aptr + k0 + 32, 0, 1);
    __builtin_prefetch(wptr + k0 + 32, 0, 1);
#if USE_ASYNC_LDS
    // gfx1250 async DMA: global -> LDS without VGPR round-trip (ASYNCcnt)
    __builtin_amdgcn_global_load_async_to_lds_b128(
        (v4i_async*)(aptr + k0), (v4i_async*)(lA + am * 32 + ak), 0, 0);
    __builtin_amdgcn_global_load_async_to_lds_b128(
        (v4i_async*)(wptr + k0), (v4i_async*)(lBT + bn * 32 + bk0), 0, 0);
    __builtin_amdgcn_s_wait_asynccnt(0);
#else
    *(uint4*)&lA[am * 32 + ak]   = *(const uint4*)(aptr + k0);
    *(uint4*)&lBT[bn * 32 + bk0] = *(const uint4*)(wptr + k0);
#endif
    __syncthreads();

    v16bf a0, a1, b0;
    const __bf16* pa0 = lA + (wm * 32 + row) * 32;
    const __bf16* pa1 = lA + (wm * 32 + 16 + row) * 32;
    const __bf16* pb  = lBT + (wn * 16 + row) * 32;
    #pragma unroll
    for (int j = 0; j < 8; ++j) {
      a0[j] = pa0[kb + j];  a0[j + 8] = pa0[16 + kb + j];
      a1[j] = pa1[kb + j];  a1[j + 8] = pa1[16 + kb + j];
      b0[j] = pb [kb + j];  b0[j + 8] = pb [16 + kb + j];
    }
    acc0 = __builtin_amdgcn_wmma_f32_16x16x32_bf16(false, a0, false, b0, (short)0, acc0, false, false);
    acc1 = __builtin_amdgcn_wmma_f32_16x16x32_bf16(false, a1, false, b0, (short)0, acc1, false, false);
    __syncthreads();
  }
}

// Grouped GEMM + bias, plain store. grid = (Npad/64, ceil(Mg/64), groups);
// weight group index = z % nw; WT group stride = K*Npad.
__global__ __launch_bounds__(256) void gemm_bias_kernel(
    const __bf16* __restrict__ A, const __bf16* __restrict__ WT, const float* __restrict__ bias,
    float* __restrict__ C, int Mg, int N, int K, int nw,
    long Astride, long Cstride, long Wstride, long BiasStride) {
  __shared__ __align__(16) __bf16 lA[64 * 32];
  __shared__ __align__(16) __bf16 lBT[64 * 32];
  int z = blockIdx.z;
  int widx = z % nw;
  const __bf16* Ag  = A + (long)z * Astride;
  const __bf16* WTg = WT + (long)widx * Wstride;
  const float*  bg  = bias + (long)widx * BiasStride;
  float*        Cg  = C + (long)z * Cstride;
  int rowBase = blockIdx.y * 64, colBase = blockIdx.x * 64;

  v8f acc0, acc1;
  gemm_core(Ag, WTg, K, rowBase, colBase, lA, lBT, acc0, acc1);

  int lane = threadIdx.x & 31, wv = threadIdx.x >> 5;
  int wm = wv & 1, wn = wv >> 1;
  int col = colBase + wn * 16 + (lane & 15);
  if (col >= N) return;
  float bv = bg[col];
  #pragma unroll
  for (int r = 0; r < 8; ++r) {
    int mr = (lane < 16) ? r : (r + 8);
    int g0 = rowBase + wm * 32 + mr;
    if (g0 < Mg) Cg[(long)g0 * N + col] = acc0[r] + bv;
    int g1 = g0 + 16;
    if (g1 < Mg) Cg[(long)g1 * N + col] = acc1[r] + bv;
  }
}

// proj GEMM with fused residual + cyclic roll(+3,+3) scatter to spatial layout.
// grid = (Npad/64, 1, BATCH*nw); group z = b*nw + w; Mg = 49.
__global__ __launch_bounds__(256) void gemm_proj_kernel(
    const __bf16* __restrict__ A, const __bf16* __restrict__ WT, const float* __restrict__ bias,
    const float* __restrict__ x_old, float* __restrict__ x_next,
    int d, int K, int nw, long Astride, long Wstride, int side, int gw) {
  __shared__ __align__(16) __bf16 lA[64 * 32];
  __shared__ __align__(16) __bf16 lBT[64 * 32];
  int z = blockIdx.z;
  int b = z / nw, w = z % nw;
  const __bf16* Ag  = A + (long)z * Astride;
  const __bf16* WTg = WT + (long)w * Wstride;
  const float*  bg  = bias + (long)w * d;
  int colBase = blockIdx.x * 64;

  v8f acc0, acc1;
  gemm_core(Ag, WTg, K, 0, colBase, lA, lBT, acc0, acc1);

  int lane = threadIdx.x & 31, wv = threadIdx.x >> 5;
  int wm = wv & 1, wn = wv >> 1;
  int col = colBase + wn * 16 + (lane & 15);
  if (col >= d) return;
  float bv = bg[col];
  int wy = w / gw, wx = w % gw;
  #pragma unroll
  for (int r = 0; r < 8; ++r) {
    int mr = (lane < 16) ? r : (r + 8);
    #pragma unroll
    for (int t = 0; t < 2; ++t) {
      int n = wm * 32 + t * 16 + mr;
      if (n >= 49) continue;
      float av = t ? acc1[r] : acc0[r];
      int y = wy * 7 + n / 7, x = wx * 7 + n % 7;
      int y2 = (y + 3) % side, x2 = (x + 3) % side;
      long src = (((long)b * side + y) * side + x) * d + col;
      long dst = (((long)b * side + y2) * side + x2) * d + col;
      x_next[dst] = x_old[src] + av + bv;
    }
  }
}

// Per-row LayerNorm with per-channel gamma/beta, in place. One wave32 per row.
__global__ __launch_bounds__(256) void ln_rows_kernel(float* __restrict__ x,
    const float* __restrict__ gg, const float* __restrict__ bb, long rows, int d) {
  int wave = threadIdx.x >> 5, lane = threadIdx.x & 31;
  long r = (long)blockIdx.x * 8 + wave;
  if (r >= rows) return;
  float* row = x + r * (long)d;
  float s = 0.f;
  for (int c = lane; c < d; c += 32) s += row[c];
  for (int o = 16; o; o >>= 1) s += __shfl_xor(s, o, 32);
  float m = s / d;
  float v = 0.f;
  for (int c = lane; c < d; c += 32) { float t = row[c] - m; v += t * t; }
  for (int o = 16; o; o >>= 1) v += __shfl_xor(v, o, 32);
  float inv = rsqrtf(v / d + 1e-5f);
  for (int c = lane; c < d; c += 32) row[c] = (row[c] - m) * inv * gg[c] + bb[c];
}

// Windowed LayerNorm: spatial x (B,side,side,d) -> h (B,nw,49,d) bf16, with
// per-window gamma/beta (nw,d). One wave32 per token.
__global__ __launch_bounds__(256) void ln_win_kernel(const float* __restrict__ x,
    const float* __restrict__ gg, const float* __restrict__ bb, __bf16* __restrict__ h,
    int side, int d, int gw, int nw) {
  int wave = threadIdx.x >> 5, lane = threadIdx.x & 31;
  long tok = (long)blockIdx.x * 8 + wave;
  long ntok = (long)BATCH * side * side;
  if (tok >= ntok) return;
  int xx = (int)(tok % side);
  long t2 = tok / side;
  int yy = (int)(t2 % side);
  int b  = (int)(t2 / side);
  const float* row = x + tok * (long)d;
  float s = 0.f;
  for (int c = lane; c < d; c += 32) s += row[c];
  for (int o = 16; o; o >>= 1) s += __shfl_xor(s, o, 32);
  float m = s / d;
  float v = 0.f;
  for (int c = lane; c < d; c += 32) { float t = row[c] - m; v += t * t; }
  for (int o = 16; o; o >>= 1) v += __shfl_xor(v, o, 32);
  float inv = rsqrtf(v / d + 1e-5f);
  int w = (yy / 7) * gw + (xx / 7);
  int n = (yy % 7) * 7 + (xx % 7);
  __bf16* out = h + (((long)b * nw + w) * 49 + n) * (long)d;
  const float* gp = gg + (long)w * d;
  const float* bp = bb + (long)w * d;
  for (int c = lane; c < d; c += 32) out[c] = tobf((row[c] - m) * inv * gp[c] + bp[c]);
}

// Attention: one block per (b*nw, head). dh = 32 at every stage. f32 VALU.
__global__ __launch_bounds__(256) void attn_kernel(const float* __restrict__ qkv,
    __bf16* __restrict__ aout, int d, int gw, int nw, int side, int layer) {
  __shared__ float qs[49 * 32], ks[49 * 32], vs[49 * 32];
  __shared__ float sc[49 * 49];
  __shared__ int ids[49];
  int z = blockIdx.x;           // b*nw + w
  int head = blockIdx.y;
  int w = z % nw;
  int tid = threadIdx.x;
  const float* base = qkv + (long)z * 49 * (3 * d) + head * 32;
  for (int i = tid; i < 49 * 32; i += 256) {
    int n = i >> 5, c = i & 31;
    const float* rowp = base + (long)n * 3 * d;
    qs[i] = rowp[c];
    ks[i] = rowp[d + c];
    vs[i] = rowp[2 * d + c];
  }
  if (tid < 49) {
    int n = tid;
    int y = (w / gw) * 7 + n / 7, x = (w % gw) * 7 + n % 7;
    int Y = (y + side * 3 - 3 * layer) % side;   // un-roll by 3*layer
    int X = (x + side * 3 - 3 * layer) % side;
    ids[tid] = (Y / 7) * gw + (X / 7);
  }
  __syncthreads();
  const float scale = 0.17677669529663687f;      // 1/sqrt(32)
  for (int p = tid; p < 49 * 49; p += 256) {
    int i = p / 49, j = p % 49;
    float s = 0.f;
    #pragma unroll 8
    for (int c = 0; c < 32; ++c) s += qs[i * 32 + c] * ks[j * 32 + c];
    sc[p] = (ids[i] == ids[j]) ? s * scale : -1e9f;
  }
  __syncthreads();
  if (tid < 49) {
    float mx = -1e30f;
    for (int j = 0; j < 49; ++j) mx = fmaxf(mx, sc[tid * 49 + j]);
    float sum = 0.f;
    for (int j = 0; j < 49; ++j) { float e = __expf(sc[tid * 49 + j] - mx); sc[tid * 49 + j] = e; sum += e; }
    float inv = 1.f / sum;
    for (int j = 0; j < 49; ++j) sc[tid * 49 + j] *= inv;
  }
  __syncthreads();
  for (int p = tid; p < 49 * 32; p += 256) {
    int i = p >> 5, c = p & 31;
    float s = 0.f;
    for (int j = 0; j < 49; ++j) s += sc[i * 49 + j] * vs[j * 32 + c];
    aout[((long)z * 49 + i) * d + head * 32 + c] = tobf(s);
  }
}

// Mean pool (B,7,7,768) -> (B,768)
__global__ void pool_kernel(const float* __restrict__ x, float* __restrict__ pooled) {
  int t = blockIdx.x * 256 + threadIdx.x;
  if (t >= BATCH * 768) return;
  int c = t % 768, b = t / 768;
  float s = 0.f;
  for (int i = 0; i < 49; ++i) s += x[((long)b * 49 + i) * 768 + c];
  pooled[t] = s * (1.f / 49.f);
}

// Classifier: (B,768) @ (768,1000) + b
__global__ void cls_kernel(const float* __restrict__ pooled, const float* __restrict__ Wc,
                           const float* __restrict__ bc, float* __restrict__ out) {
  int t = blockIdx.x * 256 + threadIdx.x;
  if (t >= BATCH * 1000) return;
  int o = t % 1000, b = t / 1000;
  float s = bc[o];
  for (int k = 0; k < 768; ++k) s += pooled[b * 768 + k] * Wc[k * 1000 + o];
  out[t] = s;
}

// ===========================================================================
extern "C" void kernel_launch(void* const* d_in, const int* in_sizes, int n_in,
                              void* d_out, int out_size, void* d_ws, size_t ws_size,
                              hipStream_t stream) {
  const int depth[4] = {2, 2, 6, 2};
  const float* feats = (const float*)d_in[0];
  const float *cls_w = nullptr, *cls_b = nullptr, *emb_w = nullptr, *emb_b = nullptr;
  const float *m_w[4], *m_b[4], *m_lng[4], *m_lnb[4];
  const float *lng[4][6], *lnb[4][6], *qw[4][6], *qb[4][6], *pw[4][6], *pb[4][6];
  auto IN = [&](int i) { return (const float*)d_in[i]; };

  if (n_in > 2 && in_sizes[1] == 1000) {
    // JAX tree_leaves order (dict keys sorted): feats; cls{b,w}; embed{b,w};
    // stages[s]{ layers[j]{ln_b,ln_g,proj_b,proj_w,qkv_b,qkv_w}, merge{b,ln_b,ln_g,w} }
    cls_b = IN(1); cls_w = IN(2); emb_b = IN(3); emb_w = IN(4);
    int idx = 5;
    for (int s = 0; s < 4; ++s) {
      for (int l = 0; l < depth[s]; ++l) {
        lnb[s][l] = IN(idx++); lng[s][l] = IN(idx++);
        pb[s][l]  = IN(idx++); pw[s][l]  = IN(idx++);
        qb[s][l]  = IN(idx++); qw[s][l]  = IN(idx++);
      }
      m_b[s] = IN(idx++); m_lnb[s] = IN(idx++); m_lng[s] = IN(idx++); m_w[s] = IN(idx++);
    }
  } else {
    // dict insertion order
    emb_w = IN(1); emb_b = IN(2);
    int idx = 3;
    for (int s = 0; s < 4; ++s) {
      m_w[s] = IN(idx++); m_b[s] = IN(idx++); m_lng[s] = IN(idx++); m_lnb[s] = IN(idx++);
      for (int l = 0; l < depth[s]; ++l) {
        lng[s][l] = IN(idx++); lnb[s][l] = IN(idx++);
        qw[s][l]  = IN(idx++); qb[s][l]  = IN(idx++);
        pw[s][l]  = IN(idx++); pb[s][l]  = IN(idx++);
      }
    }
    cls_w = IN(idx++); cls_b = IN(idx++);
  }

  // ---- workspace carve (~260 MB total); bf16 staging buffers get 256KB
  //      slack so the GEMM can read (and prefetch) unconditionally ----
  const size_t SLACK = 131072;   // bf16 elements (256 KB)
  char* w8 = (char*)d_ws;
  size_t off = 0;
  auto take = [&](size_t bytes) -> void* {
    void* p = w8 + off;
    off = (off + bytes + 255) & ~(size_t)255;
    return p;
  };
  __bf16* WcT    = (__bf16*)take((128 * 64 + SLACK) * 2);    // folded stage-0 weight, [128][64]
  float*  bc     = (float*)take(96 * 4);
  __bf16* WTbuf  = (__bf16*)take((2097152ull + SLACK) * 2);  // per-GEMM transposed weight scratch
  __bf16* Abuf   = (__bf16*)take((9633792ull + SLACK) * 2);  // max gather (stage1: 25088x384)
  float*  xA     = (float*)take(38535168ull);                // 100352 x 96 f32
  float*  xB     = (float*)take(38535168ull);
  __bf16* hbuf   = (__bf16*)take((9633792ull + SLACK) * 2);  // 100352 x 96 bf16
  float*  qkvb   = (float*)take(115605504ull);               // 100352 x 288 f32
  __bf16* aoutb  = (__bf16*)take((9633792ull + SLACK) * 2);
  float*  pooled = (float*)take(BATCH * 768 * 4);
  (void)ws_size; (void)out_size; (void)in_sizes;

  auto npad64 = [](int n) { return (n + 63) & ~63; };
  auto wtrans = [&](const float* W, int K, int N, int Npad, int nw) {
    long total = (long)K * Npad * nw;
    wtrans_kernel<<<dim3((unsigned)((total + 255) / 256)), 256, 0, stream>>>(W, WTbuf, K, N, Npad, nw);
  };

  // ---- stage 0: folded embed+merge (K padded to 64, N padded to 128) ----
  prep_w0_kernel<<<dim3((128 * 64 + 255) / 256), 256, 0, stream>>>(emb_w, emb_b, m_w[0], m_b[0], WcT, bc);
  {
    long n0 = (long)BATCH * 56 * 56 * 64;
    gather0_kernel<<<dim3((unsigned)((n0 + 255) / 256)), 256, 0, stream>>>(feats, Abuf);
  }
  {
    int M = BATCH * 56 * 56, N = 96, K = 64, Npad = 128;
    gemm_bias_kernel<<<dim3(Npad / 64, (M + 63) / 64, 1), 256, 0, stream>>>(
        Abuf, WcT, bc, xA, M, N, K, 1, 0, 0, (long)K * Npad, 0);
    ln_rows_kernel<<<dim3((M + 7) / 8), 256, 0, stream>>>(xA, m_lng[0], m_lnb[0], M, N);
  }

  float* cur = xA;
  float* oth = xB;
  for (int s = 0; s < 4; ++s) {
    int side = 56 >> s;
    int d = 96 << s;
    int gw = side / 7;
    int nw = gw * gw;
    int heads = d / 32;
    long ntok = (long)BATCH * side * side;

    for (int l = 0; l < depth[s]; ++l) {
      ln_win_kernel<<<dim3((unsigned)((ntok + 7) / 8)), 256, 0, stream>>>(
          cur, lng[s][l], lnb[s][l], hbuf, side, d, gw, nw);
      // qkv: per (b,w) GEMM 49 x d -> 49 x 3d
      {
        int N = 3 * d, K = d, Npad = npad64(N);
        wtrans(qw[s][l], K, N, Npad, nw);
        gemm_bias_kernel<<<dim3(Npad / 64, 1, BATCH * nw), 256, 0, stream>>>(
            hbuf, WTbuf, qb[s][l], qkvb,
            49, N, K, nw,
            (long)49 * d, (long)49 * N, (long)K * Npad, (long)N);
      }
      attn_kernel<<<dim3(BATCH * nw, heads), 256, 0, stream>>>(qkvb, aoutb, d, gw, nw, side, l);
      // proj + residual + roll
      {
        int N = d, K = d, Npad = npad64(N);
        wtrans(pw[s][l], K, N, Npad, nw);
        gemm_proj_kernel<<<dim3(Npad / 64, 1, BATCH * nw), 256, 0, stream>>>(
            aoutb, WTbuf, pb[s][l], cur, oth,
            d, K, nw, (long)49 * d, (long)K * Npad, side, gw);
      }
      float* t = cur; cur = oth; oth = t;
    }

    if (s < 3) {
      int s2 = side / 2;
      int T = BATCH * s2 * s2;
      int K = 4 * d, N = 2 * d, Npad = npad64(N);
      long total = (long)T * K;
      gather_merge_kernel<<<dim3((unsigned)((total + 255) / 256)), 256, 0, stream>>>(cur, Abuf, side, d);
      wtrans(m_w[s + 1], K, N, Npad, 1);
      gemm_bias_kernel<<<dim3(Npad / 64, (T + 63) / 64, 1), 256, 0, stream>>>(
          Abuf, WTbuf, m_b[s + 1], oth, T, N, K, 1, 0, 0, (long)K * Npad, 0);
      ln_rows_kernel<<<dim3((T + 7) / 8), 256, 0, stream>>>(oth, m_lng[s + 1], m_lnb[s + 1], T, N);
      float* t = cur; cur = oth; oth = t;
    }
  }

  pool_kernel<<<dim3((BATCH * 768 + 255) / 256), 256, 0, stream>>>(cur, pooled);
  cls_kernel<<<dim3((BATCH * 1000 + 255) / 256), 256, 0, stream>>>(pooled, cls_w, cls_b, (float*)d_out);
}
